// CMLA_53549652247250
// MI455X (gfx1250) — matrix operand order repaired and checked by reading, not compiled, not run on hardware
//
#include <hip/hip_runtime.h>
#include <cstdint>

#define DEVINL __device__ __forceinline__

typedef __bf16 bf16_t;
typedef __attribute__((ext_vector_type(16))) __bf16 v16bf;
typedef __attribute__((ext_vector_type(8)))  __bf16 v8bf;
typedef __attribute__((ext_vector_type(8)))  float  v8f;
typedef __attribute__((ext_vector_type(4)))  unsigned int v4u;
typedef __attribute__((ext_vector_type(8)))  int v8i;
typedef __attribute__((ext_vector_type(4)))  int v4i;

constexpr int Bc = 64, Tc = 1024, Hc = 256, Kc = 64, Uc = 128;
constexpr int U3 = 3 * Uc;                    // 384
constexpr size_t KH  = (size_t)Kc * Hc;       // 16384
constexpr size_t BTH = (size_t)Bc * Tc * Hc;  // 16.7M
constexpr size_t BTU = (size_t)Bc * Tc * Uc;  // 8.4M

// ---- WMMA operand loaders (bf16 16x16x32, wave32, contiguous k = 16*(lane/16) + s in A and B) ----
DEVINL v16bf load_A16x32(const bf16_t* A, int lda, int m0, int k0, int lane) {
  const int row = m0 + (lane & 15);
  const int kb  = k0 + ((lane & 16) ? 16 : 0);
  const bf16_t* p = A + (size_t)row * lda + kb;
  v8bf lo = *(const v8bf*)(p);
  v8bf hi = *(const v8bf*)(p + 8);
  v16bf a;
#pragma unroll
  for (int i = 0; i < 8; ++i) { a[i] = lo[i]; a[i + 8] = hi[i]; }
  return a;
}
DEVINL v16bf load_B32x16(const bf16_t* Bt, int ldb, int n0, int k0, int lane) {
  const int col = n0 + (lane & 15);
  const int kb  = k0 + ((lane & 16) ? 16 : 0);
  return *(const v16bf*)(Bt + (size_t)col * ldb + kb);
}
DEVINL v8f wmma_bf16(v16bf a, v16bf b, v8f acc) {
  return __builtin_amdgcn_wmma_f32_16x16x32_bf16(false, a, false, b,
                                                 (short)0, acc, false, false);
}
DEVINL v8f wmma_k(const bf16_t* A, int lda, const bf16_t* Bt, int ldb,
                  int m0, int n0, int Kd, int lane) {
  v8f acc = {};
  for (int k0 = 0; k0 < Kd; k0 += 32)
    acc = wmma_bf16(load_A16x32(A, lda, m0, k0, lane),
                    load_B32x16(Bt, ldb, n0, k0, lane), acc);
  return acc;
}

DEVINL float sigmoidf(float x) { return 1.0f / (1.0f + __expf(-x)); }

// ------------------------------ utility kernels ------------------------------
__global__ void k_zero(float* p, size_t n) {
  size_t i = (size_t)blockIdx.x * blockDim.x + threadIdx.x;
  if (i < n) p[i] = 0.0f;
}
__global__ void k_f2bf(const float* __restrict__ in, bf16_t* __restrict__ o, size_t n) {
  size_t i = (size_t)blockIdx.x * blockDim.x + threadIdx.x;
  if (i < n) o[i] = (bf16_t)in[i];
}
__global__ void k_initm(const float* __restrict__ m0, float* __restrict__ m) {
  int i = blockIdx.x * blockDim.x + threadIdx.x;
  if (i < Bc * Hc) m[i] = m0[i % Hc];
}
// W [Uc, U3] f32 -> Wt [U3, Uc] bf16
__global__ void k_transpose_bf(const float* __restrict__ W, bf16_t* __restrict__ Wt) {
  int i = blockIdx.x * blockDim.x + threadIdx.x;
  if (i < U3 * Uc) {
    int n = i / Uc, u = i % Uc;
    Wt[i] = (bf16_t)W[(size_t)u * U3 + n];
  }
}

// ---- einsum('khj,bj->bkh'): C[KH,B] = Ua_flat[KH,H] @ ma[B,H]^T ----
__global__ __launch_bounds__(256)
void k_tensorprod(const bf16_t* __restrict__ A, const bf16_t* __restrict__ mB,
                  bf16_t* __restrict__ outT) {
  const int lane = threadIdx.x & 31;
  const int wave = blockIdx.x * (blockDim.x >> 5) + (threadIdx.x >> 5);
  const int tilesN = Bc / 16;                   // 4
  const int mt = wave / tilesN, nt = wave % tilesN;
  v8f acc = wmma_k(A, Hc, mB, Hc, mt * 16, nt * 16, Hc, lane);
  const int n  = nt * 16 + (lane & 15);
  const int mb = mt * 16 + ((lane & 16) ? 8 : 0);
  bf16_t* p = outT + (size_t)n * KH + mb;       // transposed store
#pragma unroll
  for (int r = 0; r < 8; ++r) p[r] = (bf16_t)acc[r];
}

// ---- einsum('bth,bkh->btk') + tanh into the concat slot (bf16 [B,T,U]) ----
__global__ __launch_bounds__(256)
void k_scores(const bf16_t* __restrict__ xbf, const bf16_t* __restrict__ tmat,
              bf16_t* __restrict__ seqOut, int colOff) {
  const int b = blockIdx.y;
  const bf16_t* A  = xbf  + (size_t)b * Tc * Hc;
  const bf16_t* Bt = tmat + (size_t)b * KH;
  const int lane = threadIdx.x & 31;
  const int wave = blockIdx.x * (blockDim.x >> 5) + (threadIdx.x >> 5);
  const int tilesN = Kc / 16;                   // 4
  const int mt = wave / tilesN, nt = wave % tilesN;
  v8f acc = wmma_k(A, Hc, Bt, Hc, mt * 16, nt * 16, Hc, lane);
  const int n  = nt * 16 + (lane & 15);
  const int mb = mt * 16 + ((lane & 16) ? 8 : 0);
#pragma unroll
  for (int r = 0; r < 8; ++r)
    seqOut[((size_t)b * Tc + mb + r) * Uc + colOff + n] = (bf16_t)tanhf(acc[r]);
}

// ---- xW = seq[B*T,U] @ Wt[U3,U]^T -> f32 [B*T, 3U] ----
__global__ __launch_bounds__(256)
void k_xw(const bf16_t* __restrict__ seq, const bf16_t* __restrict__ Wt,
          float* __restrict__ xw) {
  const int lane = threadIdx.x & 31;
  const int wave = blockIdx.x * (blockDim.x >> 5) + (threadIdx.x >> 5);
  const int tilesN = U3 / 16;                   // 24
  const int mt = wave / tilesN, nt = wave % tilesN;
  v8f acc = wmma_k(seq, Uc, Wt, Uc, mt * 16, nt * 16, Uc, lane);
  const int n  = nt * 16 + (lane & 15);
  const int mb = mt * 16 + ((lane & 16) ? 8 : 0);
#pragma unroll
  for (int r = 0; r < 8; ++r)
    xw[(size_t)(mb + r) * U3 + n] = acc[r];
}

// ---- Persistent GRU scan. One workgroup (16 wave32) per GRU. h in LDS;
// R^T staged to LDS once via the Tensor Data Mover; per step each wave owns
// (row-tile, u-tile) tasks and computes z/r/h~ tiles in registers (3 WMMA
// accumulators per task), so gate math needs no LDS staging buffer. ----
__global__ __launch_bounds__(512)
void k_gru(const float* __restrict__ xwa, const float* __restrict__ xwo,
           const bf16_t* __restrict__ RaT, const bf16_t* __restrict__ RoT,
           float* __restrict__ out) {
  const float*  xw  = (blockIdx.x == 0) ? xwa : xwo;
  const bf16_t* RTg = (blockIdx.x == 0) ? RaT : RoT;
  float* o = out + (size_t)blockIdx.x * BTU;
  __shared__ bf16_t hsh[Bc * Uc];   // 16 KB
  __shared__ bf16_t RT[U3 * Uc];    // 96 KB, TDM-staged copy of R^T
  const int tid = threadIdx.x, lane = tid & 31, wave = tid >> 5;

  // --- TDM: stage R^T [U3,Uc] bf16 (98304 B = 12288 x 8B) into LDS ---
  if (wave == 0) {
    const uint64_t ga = (uint64_t)(uintptr_t)RTg;
    const uint32_t la = (uint32_t)(uintptr_t)RT;
    const uint32_t n8 = (uint32_t)((size_t)U3 * Uc * 2 / 8);  // 12288 elems
    v4u g0;
    g0[0] = 1u;                                         // count=1
    g0[1] = la;                                         // lds_addr
    g0[2] = (uint32_t)ga;                               // global_addr[31:0]
    g0[3] = (uint32_t)((ga >> 32) & 0x01FFFFFFu) | (2u << 30);  // addr hi | type=2
    v8i g1;
    g1[0] = (int)(3u << 16);      // data_size=8B, wg_mask=0, no pad/iterate
    g1[1] = (int)(n8 << 16);      // tensor_dim0[15:0] (bits 63:48)
    g1[2] = (int)(1u << 16);      // tensor_dim0 hi=0, tensor_dim1[15:0]=1
    g1[3] = (int)(n8 << 16);      // tensor_dim1 hi=0, tile_dim0=12288
    g1[4] = 0;                    // tile_dim1=0 (1-D), tile_dim2=0
    g1[5] = (int)n8;              // tensor_dim0_stride[31:0]
    g1[6] = 0;
    g1[7] = 0;
    v4i gz4 = {0, 0, 0, 0};
    v8i gz8 = {0, 0, 0, 0, 0, 0, 0, 0};
    __builtin_amdgcn_tensor_load_to_lds(g0, g1, gz4, gz4, gz8, 0);
    __builtin_amdgcn_s_wait_tensorcnt(0);
  }
  for (int i = tid; i < Bc * Uc; i += blockDim.x) hsh[i] = (bf16_t)0.0f;
  __syncthreads();

  // 32 tasks = 4 row-tiles x 8 u-tiles; 2 tasks per wave.
  const int mt0 = wave >> 3,        ug0 = wave & 7;
  const int mt1 = (wave + 16) >> 3, ug1 = (wave + 16) & 7;

  for (int t = 0; t < Tc; ++t) {
    v8f az0 = {}, ar0 = {}, ac0 = {}, az1 = {}, ar1 = {}, ac1 = {};
#pragma unroll
    for (int k0 = 0; k0 < Uc; k0 += 32) {
      v16bf a0 = load_A16x32(hsh, Uc, mt0 * 16, k0, lane);
      az0 = wmma_bf16(a0, load_B32x16(RT, Uc,          ug0 * 16, k0, lane), az0);
      ar0 = wmma_bf16(a0, load_B32x16(RT, Uc, Uc     + ug0 * 16, k0, lane), ar0);
      ac0 = wmma_bf16(a0, load_B32x16(RT, Uc, 2 * Uc + ug0 * 16, k0, lane), ac0);
      v16bf a1 = load_A16x32(hsh, Uc, mt1 * 16, k0, lane);
      az1 = wmma_bf16(a1, load_B32x16(RT, Uc,          ug1 * 16, k0, lane), az1);
      ar1 = wmma_bf16(a1, load_B32x16(RT, Uc, Uc     + ug1 * 16, k0, lane), ar1);
      ac1 = wmma_bf16(a1, load_B32x16(RT, Uc, 2 * Uc + ug1 * 16, k0, lane), ac1);
    }
    __syncthreads();   // all reads of h done before anyone writes h
#pragma unroll
    for (int task = 0; task < 2; ++task) {
      const int mt = task ? mt1 : mt0;
      const int ug = task ? ug1 : ug0;
      const v8f az = task ? az1 : az0;
      const v8f ar = task ? ar1 : ar0;
      const v8f ac = task ? ac1 : ac0;
      const int u  = ug * 16 + (lane & 15);
      const int bb = mt * 16 + ((lane & 16) ? 8 : 0);
#pragma unroll
      for (int r = 0; r < 8; ++r) {
        const int b = bb + r;
        const float* xr = xw + ((size_t)b * Tc + t) * U3;
        if (t + 1 < Tc) __builtin_prefetch(xr + U3 + u, 0, 1);
        const float z  = sigmoidf(xr[u]          + az[r]);
        const float rg = sigmoidf(xr[Uc + u]     + ar[r]);
        const float hh = tanhf   (xr[2 * Uc + u] + rg * ac[r]);
        const float h  = (float)hsh[b * Uc + u];
        const float hn = z * h + (1.0f - z) * hh;
        hsh[b * Uc + u] = (bf16_t)hn;
        o[((size_t)b * Tc + t) * Uc + u] += hn;
      }
    }
    __syncthreads();
  }
}

// ---- Attention-pooled context + memory update (layer 0 only) ----
__global__ __launch_bounds__(256)
void k_attn(const float* __restrict__ outv, const float* __restrict__ x,
            const float* __restrict__ va, const float* __restrict__ vo,
            const float* __restrict__ Ma, const float* __restrict__ Mo,
            float* __restrict__ ma, float* __restrict__ mo) {
  const int b = blockIdx.x, which = blockIdx.y;
  const float* r  = outv + (size_t)which * BTU + (size_t)b * Tc * Uc;
  const float* v  = which ? vo : va;
  const float* Mm = which ? Mo : Ma;
  float* m = (which ? mo : ma) + (size_t)b * Hc;
  __shared__ float alpha[Tc];
  __shared__ float red[256];
  __shared__ float mloc[Hc];
  const int tid = threadIdx.x;
  for (int t = tid; t < Tc; t += 256) {
    float s = 0.f;
    for (int u = 0; u < Uc; ++u) s += r[(size_t)t * Uc + u] * v[u];
    alpha[t] = s;
  }
  __syncthreads();
  float mx = -3.4e38f;
  for (int t = tid; t < Tc; t += 256) mx = fmaxf(mx, alpha[t]);
  red[tid] = mx; __syncthreads();
  for (int s = 128; s > 0; s >>= 1) { if (tid < s) red[tid] = fmaxf(red[tid], red[tid + s]); __syncthreads(); }
  const float gmax = red[0]; __syncthreads();
  float psum = 0.f;
  for (int t = tid; t < Tc; t += 256) { float e = __expf(alpha[t] - gmax); alpha[t] = e; psum += e; }
  red[tid] = psum; __syncthreads();
  for (int s = 128; s > 0; s >>= 1) { if (tid < s) red[tid] += red[tid + s]; __syncthreads(); }
  const float inv = 1.0f / red[0]; __syncthreads();
  for (int t = tid; t < Tc; t += 256) alpha[t] *= inv;
  for (int h = tid; h < Hc; h += 256) mloc[h] = m[h];
  __syncthreads();
  for (int h = tid; h < Hc; h += 256) {
    float c = 0.f;
    for (int t = 0; t < Tc; ++t) c += alpha[t] * x[((size_t)b * Tc + t) * Hc + h];
    float mm = 0.f;
    for (int j = 0; j < Hc; ++j) mm += mloc[j] * Mm[(size_t)j * Hc + h];
    m[h] = tanhf(mm) + c;
  }
}

extern "C" void kernel_launch(void* const* d_in, const int* in_sizes, int n_in,
                              void* d_out, int out_size, void* d_ws, size_t ws_size,
                              hipStream_t stream) {
  (void)in_sizes; (void)n_in; (void)out_size; (void)ws_size;
  const float* x   = (const float*)d_in[0];
  const float* m0a = (const float*)d_in[1];
  const float* m0o = (const float*)d_in[2];
  const float* Ua  = (const float*)d_in[3];
  const float* Uo  = (const float*)d_in[4];
  const float* Va  = (const float*)d_in[5];
  const float* Vo  = (const float*)d_in[6];
  const float* Ma  = (const float*)d_in[7];
  const float* Mo  = (const float*)d_in[8];
  const float* va  = (const float*)d_in[9];
  const float* vo  = (const float*)d_in[10];
  const float* Wa  = (const float*)d_in[11];
  const float* Ra  = (const float*)d_in[12];
  const float* Wo  = (const float*)d_in[13];
  const float* Ro  = (const float*)d_in[14];
  float* out = (float*)d_out;

  char* ws = (char*)d_ws;
  size_t off = 0;
  auto take = [&](size_t bytes) -> void* {
    void* p = ws + off; off = (off + bytes + 255) & ~(size_t)255; return p;
  };
  bf16_t* xbf  = (bf16_t*)take(BTH * 2);
  const float* Usrc[4] = {Ua, Uo, Va, Vo};
  bf16_t* Ubf[4];
  for (int i = 0; i < 4; ++i) Ubf[i] = (bf16_t*)take(KH * Hc * 2);
  bf16_t* WaT = (bf16_t*)take((size_t)U3 * Uc * 2);
  bf16_t* RaT = (bf16_t*)take((size_t)U3 * Uc * 2);
  bf16_t* WoT = (bf16_t*)take((size_t)U3 * Uc * 2);
  bf16_t* RoT = (bf16_t*)take((size_t)U3 * Uc * 2);
  float*  ma   = (float*)take((size_t)Bc * Hc * 4);
  float*  mo   = (float*)take((size_t)Bc * Hc * 4);
  bf16_t* mabf = (bf16_t*)take((size_t)Bc * Hc * 2);
  bf16_t* mobf = (bf16_t*)take((size_t)Bc * Hc * 2);
  bf16_t* ta   = (bf16_t*)take((size_t)Bc * KH * 2);
  bf16_t* tva  = (bf16_t*)take((size_t)Bc * KH * 2);
  bf16_t* toa  = (bf16_t*)take((size_t)Bc * KH * 2);
  bf16_t* tvo  = (bf16_t*)take((size_t)Bc * KH * 2);
  bf16_t* asp  = (bf16_t*)take(BTU * 2);
  bf16_t* opi  = (bf16_t*)take(BTU * 2);
  float*  xwa  = (float*)take(BTU * 3 * 4);
  float*  xwo  = (float*)take(BTU * 3 * 4);

  const int TPB = 256;
  const size_t nout = 2 * BTU;
  k_zero<<<(unsigned)((nout + TPB - 1) / TPB), TPB, 0, stream>>>(out, nout);
  k_initm<<<(Bc * Hc + TPB - 1) / TPB, TPB, 0, stream>>>(m0a, ma);
  k_initm<<<(Bc * Hc + TPB - 1) / TPB, TPB, 0, stream>>>(m0o, mo);
  k_f2bf<<<(unsigned)((BTH + TPB - 1) / TPB), TPB, 0, stream>>>(x, xbf, BTH);
  for (int i = 0; i < 4; ++i)
    k_f2bf<<<(unsigned)((KH * Hc + TPB - 1) / TPB), TPB, 0, stream>>>(Usrc[i], Ubf[i], KH * Hc);
  k_transpose_bf<<<(U3 * Uc + TPB - 1) / TPB, TPB, 0, stream>>>(Wa, WaT);
  k_transpose_bf<<<(U3 * Uc + TPB - 1) / TPB, TPB, 0, stream>>>(Ra, RaT);
  k_transpose_bf<<<(U3 * Uc + TPB - 1) / TPB, TPB, 0, stream>>>(Wo, WoT);
  k_transpose_bf<<<(U3 * Uc + TPB - 1) / TPB, TPB, 0, stream>>>(Ro, RoT);

  for (int layer = 0; layer < 2; ++layer) {
    k_f2bf<<<(Bc * Hc + TPB - 1) / TPB, TPB, 0, stream>>>(ma, mabf, (size_t)Bc * Hc);
    k_f2bf<<<(Bc * Hc + TPB - 1) / TPB, TPB, 0, stream>>>(mo, mobf, (size_t)Bc * Hc);
    k_tensorprod<<<512, 256, 0, stream>>>(Ubf[0], mabf, ta);   // Ua . ma
    k_tensorprod<<<512, 256, 0, stream>>>(Ubf[2], mobf, tva);  // Va . mo
    k_tensorprod<<<512, 256, 0, stream>>>(Ubf[1], mabf, toa);  // Uo . ma
    k_tensorprod<<<512, 256, 0, stream>>>(Ubf[3], mobf, tvo);  // Vo . mo
    dim3 sg(32, Bc);
    k_scores<<<sg, 256, 0, stream>>>(xbf, ta,  asp, 0);
    k_scores<<<sg, 256, 0, stream>>>(xbf, tva, asp, Kc);
    k_scores<<<sg, 256, 0, stream>>>(xbf, toa, opi, 0);
    k_scores<<<sg, 256, 0, stream>>>(xbf, tvo, opi, Kc);
    k_xw<<<12288, 256, 0, stream>>>(asp, WaT, xwa);
    k_xw<<<12288, 256, 0, stream>>>(opi, WoT, xwo);
    k_gru<<<2, 512, 0, stream>>>(xwa, xwo, RaT, RoT, out);
    if (layer == 0) {
      dim3 ag(Bc, 2);
      k_attn<<<ag, 256, 0, stream>>>(out, x, va, vo, Ma, Mo, ma, mo);
    }
  }
}
